// MAB_52828097741394
// MI455X (gfx1250) — matrix operand order repaired and checked by reading, not compiled, not run on hardware
//
#include <hip/hip_runtime.h>
#include <hip/hip_bf16.h>

typedef __attribute__((ext_vector_type(16))) _Float16 v16h;
typedef __attribute__((ext_vector_type(8)))  _Float16 v8h;
typedef __attribute__((ext_vector_type(8)))  float    v8f;

__device__ __forceinline__ v8f wmma_f16(v16h a, v16h b, v8f c) {
  // D = A(16x32 f16) * B(32x16 f16) + C(16x16 f32)
  return __builtin_amdgcn_wmma_f32_16x16x32_f16(
      /*neg_a=*/false, a, /*neg_b=*/false, b,
      /*c_mod=*/(short)0, c, /*reuse_a=*/false, /*reuse_b=*/false);
}

__device__ __forceinline__ v8f zero8() {
  v8f z;
  for (int i = 0; i < 8; ++i) z[i] = 0.f;
  return z;
}

union FragH { v16h v; v8h h[2]; };

// ---- CDNA5 async copy: global -> LDS, 16 bytes per lane, ASYNCcnt tracked --
__device__ __forceinline__ void async_cp16(const _Float16* gsrc, _Float16* ldst) {
  unsigned lofs = (unsigned)(unsigned long long)(const void*)ldst;  // LDS byte offset
  unsigned long long ga = (unsigned long long)(const void*)gsrc;
  asm volatile("global_load_async_to_lds_b128 %0, %1, off"
               :: "v"(lofs), "v"(ga) : "memory");
}
__device__ __forceinline__ void wait_async0() {
  asm volatile("s_wait_asynccnt 0" ::: "memory");
}

// ---------------------------------------------------------------------------
// Pre-pass conversions (run once, tiny vs. ~34 GFLOP of matmul):
//   cvt_f16:   Y[i] = (f16) X[i]
//   cvt_f16_t: Wt[n*K + k] = (f16) W[k*N + n]   (512x512)
// ---------------------------------------------------------------------------
__global__ __launch_bounds__(256) void cvt_f16_kernel(
    const float* __restrict__ X, _Float16* __restrict__ Y) {
  size_t i = ((size_t)blockIdx.x * 256 + threadIdx.x) * 4;
  float4 v = *(const float4*)(X + i);
  Y[i + 0] = (_Float16)v.x;
  Y[i + 1] = (_Float16)v.y;
  Y[i + 2] = (_Float16)v.z;
  Y[i + 3] = (_Float16)v.w;
}

__global__ __launch_bounds__(256) void cvt_f16_t_kernel(
    const float* __restrict__ W, _Float16* __restrict__ Wt) {
  int i = blockIdx.x * 256 + threadIdx.x;   // 0 .. 512*512-1
  int k = i >> 9, n = i & 511;
  Wt[(size_t)n * 512 + k] = (_Float16)W[i];
}

// ---------------------------------------------------------------------------
// Tiled GEMM (f16 in, WMMA f32 accum):
//   A:  [Z, M, Kd] f16 row-major     Wt: [N, Kd] f16 (pre-transposed)
//   mode 0: Cf  = A@W + bias                       (f32 out)
//   mode 1: Ch  = A@W + bias                       (f16 out, row-major)
//   mode 2: ChT = A@W + bias, stored [Z, N, M]     (f16 out, transposed)
//   mode 3: Cf  = resid + relu(A@W + bias)         (f32 out)
// 256 threads / 8 waves; 128x128 tile; K-step 32; double-buffered async LDS.
// ---------------------------------------------------------------------------
__global__ __launch_bounds__(256) void gemm_f16_kernel(
    const _Float16* __restrict__ A, const _Float16* __restrict__ Wt,
    const float* __restrict__ bias, const float* __restrict__ resid,
    float* __restrict__ Cf, _Float16* __restrict__ Ch,
    int M, int Kd, int N, int mode) {
  __shared__ _Float16 As[2][128][32];
  __shared__ _Float16 Bs[2][128][32];

  const int tid  = threadIdx.x;
  const int w    = tid >> 5;
  const int lane = tid & 31;
  const int hi   = lane >> 4;
  const int ln   = lane & 15;
  const int wm   = w >> 1;
  const int wn   = w & 1;

  const int n0 = blockIdx.x * 128;
  const int m0 = blockIdx.y * 128;
  const size_t zA = (size_t)blockIdx.z * (size_t)M * (size_t)Kd;
  const size_t zC = (size_t)blockIdx.z * (size_t)M * (size_t)N;

  const _Float16* Abase = A + zA + (size_t)m0 * Kd;
  const _Float16* Wbase = Wt + (size_t)n0 * Kd;

  v8f acc[2][4];
  for (int mi = 0; mi < 2; ++mi)
    for (int ni = 0; ni < 4; ++ni) acc[mi][ni] = zero8();

  const int nk = Kd / 32;

  // issue async tile loads for K-step `s` into buffer `buf` (branch-free)
  auto issue = [&](int s, int buf) {
    const int kb = s * 32;
    for (int i = 0; i < 2; ++i) {           // A tile: 512 x 16B chunks
      int cid = i * 256 + tid;
      int r = cid >> 2, ko = (cid & 3) * 8;
      async_cp16(Abase + (size_t)r * Kd + kb + ko, &As[buf][r][ko]);
    }
    for (int i = 0; i < 2; ++i) {           // W tile: 512 x 16B chunks
      int cid = i * 256 + tid;
      int n = cid >> 2, ko = (cid & 3) * 8;
      async_cp16(Wbase + (size_t)n * Kd + kb + ko, &Bs[buf][n][ko]);
    }
  };

  issue(0, 0);
  for (int s = 0; s < nk; ++s) {
    const int buf = s & 1;
    wait_async0();
    __syncthreads();                 // tile s landed; tile s-1 readers done
    if (s + 1 < nk) issue(s + 1, buf ^ 1);

    FragH a[2], bf[4];
    for (int mi = 0; mi < 2; ++mi) {
      const _Float16* ar = &As[buf][wm * 32 + mi * 16 + ln][0];
      a[mi].h[0] = *(const v8h*)(ar + hi * 8);
      a[mi].h[1] = *(const v8h*)(ar + 16 + hi * 8);
    }
    for (int ni = 0; ni < 4; ++ni) {
      const _Float16* br = &Bs[buf][wn * 64 + ni * 16 + ln][0];
      bf[ni].h[0] = *(const v8h*)(br + hi * 8);
      bf[ni].h[1] = *(const v8h*)(br + 16 + hi * 8);
    }
    for (int mi = 0; mi < 2; ++mi)
      for (int ni = 0; ni < 4; ++ni)
        acc[mi][ni] = wmma_f16(a[mi].v, bf[ni].v, acc[mi][ni]);
  }

  // ---- epilogue ----
  for (int mi = 0; mi < 2; ++mi)
    for (int ni = 0; ni < 4; ++ni) {
      int cn = n0 + wn * 64 + ni * 16 + ln;
      float bv = bias[cn];
      for (int v = 0; v < 8; ++v) {
        int cm = m0 + wm * 32 + mi * 16 + v + 8 * hi;   // D row = v + 8*hi
        float val = acc[mi][ni][v] + bv;
        if (mode == 0) {
          Cf[zC + (size_t)cm * N + cn] = val;
        } else if (mode == 1) {
          Ch[zC + (size_t)cm * N + cn] = (_Float16)val;
        } else if (mode == 2) {
          Ch[zC + (size_t)cn * M + cm] = (_Float16)val;
        } else {
          size_t idx = zC + (size_t)cm * N + cn;
          Cf[idx] = resid[idx] + fmaxf(val, 0.f);
        }
      }
    }
}

// ---------------------------------------------------------------------------
// Flash attention + residual:
//   Ot[b,q,h*64+d] = Qp[b,q,h*64+d] + softmax(Qh.Khᵀ/sqrt(512)) @ Vh
//   Kph: [B, NK, 512] f16 ; VpT: [B, 512, NK] f16 (head-dim major)
// 256 threads (8 waves) per (b, h, 128-q block); wave owns 16 q rows.
// Double-buffered async LDS staging of K / Vᵀ tiles.
// ---------------------------------------------------------------------------
__global__ __launch_bounds__(256) void attn_kernel(
    const float* __restrict__ Qp, const _Float16* __restrict__ Kph,
    const _Float16* __restrict__ VpT, float* __restrict__ Ot) {
  const int qb = blockIdx.x;
  const int h  = blockIdx.y;
  const int b  = blockIdx.z;
  const int tid  = threadIdx.x;
  const int w    = tid >> 5;
  const int lane = tid & 31;
  const int hi   = lane >> 4;
  const int ln   = lane & 15;

  __shared__ _Float16 Ks[2][128][64];    // keys  [k][d]          2x16 KB
  __shared__ _Float16 Vs[2][64][128];    // V transposed [d][k]   2x16 KB
  __shared__ _Float16 Ps[8][16][128];    // per-wave P staging      32 KB

  const float scale = 0.044194173824159216f;  // 1/sqrt(512)
  const int qrow0 = qb * 128 + w * 16;

  // ---- Q A-fragments (16 rows x 64 dims -> two K=32 fragments), once ----
  v16h qa[2];
  {
    const float* qbase = Qp + ((size_t)(b * 1024 + qrow0 + ln)) * 512 + h * 64;
    for (int c = 0; c < 2; ++c)
      for (int t = 0; t < 8; ++t) {
        qa[c][t]     = (_Float16)(qbase[c * 32 + hi * 8 + t]);
        qa[c][t + 8] = (_Float16)(qbase[c * 32 + 16 + hi * 8 + t]);
      }
  }

  v8f o[4];
  for (int ot = 0; ot < 4; ++ot) o[ot] = zero8();
  float mrun[8], lrun[8];
  for (int v = 0; v < 8; ++v) { mrun[v] = -3.0e38f; lrun[v] = 0.f; }

  // issue async K/Vᵀ tile loads for key-block kb into buffer buf (branch-free)
  auto issue = [&](int kb, int buf) {
    for (int i = 0; i < 4; ++i) {          // K tile: 1024 x 16B chunks
      int cid = i * 256 + tid;
      int k = cid >> 3, doff = (cid & 7) * 8;
      async_cp16(Kph + ((size_t)(b * 1024 + kb * 128 + k)) * 512 + h * 64 + doff,
                 &Ks[buf][k][doff]);
    }
    for (int i = 0; i < 4; ++i) {          // Vᵀ tile: 1024 x 16B chunks
      int cid = i * 256 + tid;
      int d = cid >> 4, koff = (cid & 15) * 8;
      async_cp16(VpT + ((size_t)(b * 512 + h * 64 + d)) * 1024 + kb * 128 + koff,
                 &Vs[buf][d][koff]);
    }
  };

  issue(0, 0);
  for (int kb = 0; kb < 8; ++kb) {
    const int buf = kb & 1;
    wait_async0();
    __syncthreads();                 // tile kb landed; tile kb-1 readers done
    if (kb + 1 < 8) issue(kb + 1, buf ^ 1);

    // ---- S = Q @ Kᵀ : 8 tiles of 16x16, K=64 as two chained WMMAs ----
    v8f s[8];
    for (int nt = 0; nt < 8; ++nt) {
      const _Float16* kr = &Ks[buf][nt * 16 + ln][0];
      FragH b0, b1;
      b0.h[0] = *(const v8h*)(kr + hi * 8);
      b0.h[1] = *(const v8h*)(kr + 16 + hi * 8);
      b1.h[0] = *(const v8h*)(kr + 32 + hi * 8);
      b1.h[1] = *(const v8h*)(kr + 32 + 16 + hi * 8);
      v8f t = wmma_f16(qa[0], b0.v, zero8());
      s[nt]  = wmma_f16(qa[1], b1.v, t);
    }

    // ---- online softmax (row = v + 8*hi; reductions within 16-lane half) --
    float corr[8];
    for (int v = 0; v < 8; ++v) {
      float mx = -3.0e38f;
      for (int nt = 0; nt < 8; ++nt) {
        s[nt][v] *= scale;
        mx = fmaxf(mx, s[nt][v]);
      }
      for (int msk = 1; msk < 16; msk <<= 1)
        mx = fmaxf(mx, __shfl_xor(mx, msk, 32));
      float mnew = fmaxf(mrun[v], mx);
      corr[v] = __expf(mrun[v] - mnew);
      mrun[v] = mnew;
      float ps = 0.f;
      for (int nt = 0; nt < 8; ++nt) {
        float p = __expf(s[nt][v] - mnew);
        s[nt][v] = p;
        ps += p;
      }
      for (int msk = 1; msk < 16; msk <<= 1)
        ps += __shfl_xor(ps, msk, 32);
      lrun[v] = lrun[v] * corr[v] + ps;
    }
    for (int ot = 0; ot < 4; ++ot)
      for (int v = 0; v < 8; ++v) o[ot][v] *= corr[v];

    // ---- stage P (D-layout) into LDS, reload as A-fragments ----
    for (int nt = 0; nt < 8; ++nt)
      for (int v = 0; v < 8; ++v)
        Ps[w][v + 8 * hi][nt * 16 + ln] = (_Float16)s[nt][v];
    // same-wave DS ops are in-order: safe to reload without a block barrier
    FragH pa[4];
    {
      const _Float16* pr = &Ps[w][ln][0];
      for (int c = 0; c < 4; ++c) {
        pa[c].h[0] = *(const v8h*)(pr + c * 32 + hi * 8);
        pa[c].h[1] = *(const v8h*)(pr + c * 32 + 16 + hi * 8);
      }
    }
    // ---- O += P @ V : 4 output tiles x 4 K-chunks ----
    for (int ot = 0; ot < 4; ++ot) {
      const _Float16* vr = &Vs[buf][ot * 16 + ln][0];
      for (int c = 0; c < 4; ++c) {
        FragH vb;
        vb.h[0] = *(const v8h*)(vr + c * 32 + hi * 8);
        vb.h[1] = *(const v8h*)(vr + c * 32 + 16 + hi * 8);
        o[ot] = wmma_f16(pa[c].v, vb.v, o[ot]);
      }
    }
  }

  // ---- epilogue: Ot = Qp + O / l ----
  for (int v = 0; v < 8; ++v) {
    float inv = 1.0f / lrun[v];
    int q = qrow0 + v + 8 * hi;
    for (int ot = 0; ot < 4; ++ot) {
      int dcol = h * 64 + ot * 16 + ln;
      size_t idx = ((size_t)(b * 1024 + q)) * 512 + dcol;
      Ot[idx] = Qp[idx] + o[ot][v] * inv;
    }
  }
}

// ---------------------------------------------------------------------------
// LayerNorm over last dim (512), one block (256 thr) per row.
// Optionally also writes an f16 copy (A operand of the next GEMM).
// ---------------------------------------------------------------------------
__global__ __launch_bounds__(256) void layernorm_kernel(
    const float* __restrict__ X, const float* __restrict__ g,
    const float* __restrict__ bt, float* __restrict__ Y,
    _Float16* __restrict__ Yh) {
  const int row = blockIdx.x;
  const int tid = threadIdx.x;
  const float* x = X + (size_t)row * 512;
  float v0 = x[tid];
  float v1 = x[tid + 256];
  float s1 = v0 + v1;
  float s2 = v0 * v0 + v1 * v1;
  for (int msk = 1; msk < 32; msk <<= 1) {
    s1 += __shfl_xor(s1, msk, 32);
    s2 += __shfl_xor(s2, msk, 32);
  }
  __shared__ float red1[8], red2[8];
  if ((tid & 31) == 0) { red1[tid >> 5] = s1; red2[tid >> 5] = s2; }
  __syncthreads();
  float t1 = 0.f, t2 = 0.f;
  for (int i = 0; i < 8; ++i) { t1 += red1[i]; t2 += red2[i]; }
  float mean = t1 * (1.0f / 512.0f);
  float var  = t2 * (1.0f / 512.0f) - mean * mean;
  float r = rsqrtf(var + 1e-5f);
  float y0 = (v0 - mean) * r * g[tid] + bt[tid];
  float y1 = (v1 - mean) * r * g[tid + 256] + bt[tid + 256];
  Y[(size_t)row * 512 + tid]       = y0;
  Y[(size_t)row * 512 + tid + 256] = y1;
  if (Yh) {
    Yh[(size_t)row * 512 + tid]       = (_Float16)y0;
    Yh[(size_t)row * 512 + tid + 256] = (_Float16)y1;
  }
}

extern "C" void kernel_launch(void* const* d_in, const int* in_sizes, int n_in,
                              void* d_out, int out_size, void* d_ws, size_t ws_size,
                              hipStream_t stream) {
  const float* Q  = (const float*)d_in[0];
  const float* K  = (const float*)d_in[1];
  const float* Wq = (const float*)d_in[2];
  const float* bq = (const float*)d_in[3];
  const float* Wk = (const float*)d_in[4];
  const float* bk = (const float*)d_in[5];
  const float* Wv = (const float*)d_in[6];
  const float* bv = (const float*)d_in[7];
  const float* Wo = (const float*)d_in[8];
  const float* bo = (const float*)d_in[9];
  const float* g0 = (const float*)d_in[10];
  const float* b0 = (const float*)d_in[11];
  const float* g1 = (const float*)d_in[12];
  const float* b1 = (const float*)d_in[13];
  float* out = (float*)d_out;

  const size_t T32 = (size_t)8 * 1024 * 512;  // elements of a [B,N,512] tensor
  const size_t W16 = (size_t)512 * 512;       // elements of a 512x512 weight

  // workspace carve-up
  char* base = (char*)d_ws;
  float*    Qp   = (float*)base;                     base += T32 * 4;  // 16 MB
  float*    Ot   = (float*)base;                     base += T32 * 4;  // 16 MB
  float*    O0   = (float*)base;                     base += T32 * 4;  // 16 MB
  _Float16* Qh   = (_Float16*)base;                  base += T32 * 2;  //  8 MB
  _Float16* Kh   = (_Float16*)base;                  base += T32 * 2;  //  8 MB
  _Float16* Kph  = (_Float16*)base;                  base += T32 * 2;  //  8 MB
  _Float16* VpT  = (_Float16*)base;                  base += T32 * 2;  //  8 MB
  _Float16* O0h  = (_Float16*)base;                  base += T32 * 2;  //  8 MB
  _Float16* Wqt  = (_Float16*)base;                  base += W16 * 2;
  _Float16* Wkt  = (_Float16*)base;                  base += W16 * 2;
  _Float16* Wvt  = (_Float16*)base;                  base += W16 * 2;
  _Float16* Wot  = (_Float16*)base;                  base += W16 * 2;

  dim3 bb(256);
  // ---- pre-pass: f16 copies / transposed f16 weights ----
  cvt_f16_kernel<<<dim3(T32 / 1024), bb, 0, stream>>>(Q, Qh);
  cvt_f16_kernel<<<dim3(T32 / 1024), bb, 0, stream>>>(K, Kh);
  cvt_f16_t_kernel<<<dim3(W16 / 256), bb, 0, stream>>>(Wq, Wqt);
  cvt_f16_t_kernel<<<dim3(W16 / 256), bb, 0, stream>>>(Wk, Wkt);
  cvt_f16_t_kernel<<<dim3(W16 / 256), bb, 0, stream>>>(Wv, Wvt);
  cvt_f16_t_kernel<<<dim3(W16 / 256), bb, 0, stream>>>(Wo, Wot);

  dim3 gg(4, 8, 8);  // N/128, M/128, B
  // ---- projections ----
  gemm_f16_kernel<<<gg, bb, 0, stream>>>(Qh, Wqt, bq, nullptr, Qp, nullptr,
                                         1024, 512, 512, /*mode=*/0);
  gemm_f16_kernel<<<gg, bb, 0, stream>>>(Kh, Wkt, bk, nullptr, nullptr, Kph,
                                         1024, 512, 512, /*mode=*/1);
  gemm_f16_kernel<<<gg, bb, 0, stream>>>(Kh, Wvt, bv, nullptr, nullptr, VpT,
                                         1024, 512, 512, /*mode=*/2);
  // ---- attention + residual ----
  attn_kernel<<<dim3(8, 8, 8), bb, 0, stream>>>(Qp, Kph, VpT, Ot);
  // ---- LN0 ----
  layernorm_kernel<<<dim3(8192), bb, 0, stream>>>(Ot, g0, b0, O0, O0h);
  // ---- O = O0 + relu(O0 @ Wo + bo) ----
  gemm_f16_kernel<<<gg, bb, 0, stream>>>(O0h, Wot, bo, O0, Ot, nullptr,
                                         1024, 512, 512, /*mode=*/3);
  // ---- LN1 -> final output ----
  layernorm_kernel<<<dim3(8192), bb, 0, stream>>>(Ot, g1, b1, out, nullptr);
}
